// WeightedContrastiveLoss_50173807951910
// MI455X (gfx1250) — compile-verified, matrix operand order checked by reading, not run on hardware
//
#include <hip/hip_runtime.h>
#include <hip/hip_bf16.h>
#include <stdint.h>

// ---------------------------------------------------------------------------
// Types for CDNA5 WMMA
// ---------------------------------------------------------------------------
typedef __attribute__((ext_vector_type(16))) __bf16 v16bf;
typedef __attribute__((ext_vector_type(8)))  __bf16 v8bf;
typedef __attribute__((ext_vector_type(8)))  float  v8f;
typedef __attribute__((ext_vector_type(4)))  int    v4i;

#define B_ROWS 4096
#define D_DIM  2048
#define BN_EPS 1e-5f
#define INV_TEMP 10.0f

// Async global->LDS staging (gfx1250 ASYNCcnt path), guarded so the file
// still compiles on toolchains without the builtins.
#if defined(__has_builtin)
#if __has_builtin(__builtin_amdgcn_global_load_async_to_lds_b128) && \
    __has_builtin(__builtin_amdgcn_s_wait_asynccnt)
#define USE_ASYNC_LDS 1
#endif
#endif
#ifndef USE_ASYNC_LDS
#define USE_ASYNC_LDS 0
#endif

#define AS_GLOBAL __attribute__((address_space(1)))
#define AS_SHARED __attribute__((address_space(3)))
typedef AS_GLOBAL v4i* v4i_gp;   // pointer to AS1 int4 (16B global chunk)
typedef AS_SHARED v4i* v4i_sp;   // pointer to AS3 int4 (16B LDS chunk)

// ---------------------------------------------------------------------------
// f32 -> bf16 convert
// ---------------------------------------------------------------------------
__global__ __launch_bounds__(256)
void cvt_bf16_kernel(const float* __restrict__ x, __bf16* __restrict__ y, int n) {
    int i = blockIdx.x * 256 + threadIdx.x;
    if (i < n) y[i] = (__bf16)x[i];
}

// ---------------------------------------------------------------------------
// Tiled bf16 WMMA GEMM:  C[M,N] = A[M,K] * Bm[N,K]^T  (+ optional bias[N])
// Block tile 128x256x32, 256 threads = 8 waves, wave tile 64x64 (4x4 WMMAs).
// Double-buffered LDS, async global->LDS staging when available.
// ---------------------------------------------------------------------------
#define GBM 128
#define GBN 256
#define GBK 32
#define LDSK 40   // padded row stride (elements); 40*2B = 80B, 16B aligned

__global__ __launch_bounds__(256)
void gemm_nt_bf16_kernel(const __bf16* __restrict__ A,
                         const __bf16* __restrict__ Bm,
                         float* __restrict__ C,
                         const float* __restrict__ bias,
                         int M, int N, int K) {
    __shared__ __align__(16) __bf16 As[2][GBM * LDSK];
    __shared__ __align__(16) __bf16 Bs[2][GBN * LDSK];

    const int tid  = threadIdx.x;
    const int wave = tid >> 5;
    const int lane = tid & 31;
    const int bm = blockIdx.y * GBM;
    const int bn = blockIdx.x * GBN;

    // wave tiling: 2 waves along M (64 rows each), 4 waves along N (64 cols)
    const int wm = (wave & 1) * 64;
    const int wn = (wave >> 1) * 64;

    // WMMA fragment lane mapping (CDNA5 16-bit A/B layout):
    // lane&15 = matrix row (A) / col (B); (lane>>4)*8 = K chunk base,
    // chunks at [fk, fk+8) and [fk+16, fk+24).
    const int fr = lane & 15;
    const int fk = (lane >> 4) * 8;

    v8f zero = {};
    v8f acc[4][4];
#pragma unroll
    for (int i = 0; i < 4; ++i)
#pragma unroll
        for (int j = 0; j < 4; ++j) acc[i][j] = zero;

    union Frag { v16bf f; v8bf h[2]; };

    // loader mapping:
    //   A tile (128x32): thread t -> row t>>1, k-half (t&1)*16  (32B each)
    //   B tile (256x32): thread t -> row t, full 32-k row       (64B each)
    const int arow = tid >> 1;
    const int ak0  = (tid & 1) * 16;
    const int brow = tid;

#if USE_ASYNC_LDS
    auto stage_async = [&](int kb, int b) {
        const __bf16* ga = A + (size_t)(bm + arow) * K + kb + ak0;
        __bf16* la = &As[b][arow * LDSK + ak0];
        __builtin_amdgcn_global_load_async_to_lds_b128(
            (v4i_gp)(uintptr_t)(ga), (v4i_sp)(uintptr_t)(la), 0, 0);
        __builtin_amdgcn_global_load_async_to_lds_b128(
            (v4i_gp)(uintptr_t)(ga + 8), (v4i_sp)(uintptr_t)(la + 8), 0, 0);

        const __bf16* gb = Bm + (size_t)(bn + brow) * K + kb;
        __bf16* lb = &Bs[b][brow * LDSK];
#pragma unroll
        for (int c = 0; c < 4; ++c)
            __builtin_amdgcn_global_load_async_to_lds_b128(
                (v4i_gp)(uintptr_t)(gb + 8 * c), (v4i_sp)(uintptr_t)(lb + 8 * c), 0, 0);
    };
#else
    auto stage_regs = [&](int kb) {
        const __bf16* ga = A + (size_t)(bm + arow) * K + kb + ak0;
        v8bf a0 = *(const v8bf*)(ga);
        v8bf a1 = *(const v8bf*)(ga + 8);
        *(v8bf*)(&As[0][arow * LDSK + ak0])     = a0;
        *(v8bf*)(&As[0][arow * LDSK + ak0 + 8]) = a1;

        const __bf16* gb = Bm + (size_t)(bn + brow) * K + kb;
        v8bf b0 = *(const v8bf*)(gb);
        v8bf b1 = *(const v8bf*)(gb + 8);
        v8bf b2 = *(const v8bf*)(gb + 16);
        v8bf b3 = *(const v8bf*)(gb + 24);
        *(v8bf*)(&Bs[0][brow * LDSK])      = b0;
        *(v8bf*)(&Bs[0][brow * LDSK + 8])  = b1;
        *(v8bf*)(&Bs[0][brow * LDSK + 16]) = b2;
        *(v8bf*)(&Bs[0][brow * LDSK + 24]) = b3;
    };
#endif

    auto compute = [&](int b) {
        Frag af[4], bf[4];
#pragma unroll
        for (int i = 0; i < 4; ++i) {
            const __bf16* p = &As[b][(wm + i * 16 + fr) * LDSK + fk];
            af[i].h[0] = *(const v8bf*)(p);
            af[i].h[1] = *(const v8bf*)(p + 16);
        }
#pragma unroll
        for (int j = 0; j < 4; ++j) {
            const __bf16* p = &Bs[b][(wn + j * 16 + fr) * LDSK + fk];
            bf[j].h[0] = *(const v8bf*)(p);
            bf[j].h[1] = *(const v8bf*)(p + 16);
        }
#pragma unroll
        for (int i = 0; i < 4; ++i)
#pragma unroll
            for (int j = 0; j < 4; ++j)
                acc[i][j] = __builtin_amdgcn_wmma_f32_16x16x32_bf16(
                    false, af[i].f, false, bf[j].f,
                    (short)0, acc[i][j], false, false);
    };

#if USE_ASYNC_LDS
    const int nk = K / GBK;
    stage_async(0, 0);
    __builtin_amdgcn_s_wait_asynccnt(0);
    __syncthreads();
    for (int s = 0; s < nk; ++s) {
        const int cur = s & 1;
        if (s + 1 < nk) stage_async((s + 1) * GBK, cur ^ 1);
        compute(cur);
        __builtin_amdgcn_s_wait_asynccnt(0);
        __syncthreads();
    }
#else
    for (int kb = 0; kb < K; kb += GBK) {
        stage_regs(kb);
        __syncthreads();
        compute(0);
        __syncthreads();
    }
#endif

    // ---- epilogue: C/D layout lanes0-15 -> M=v, lanes16-31 -> M=8+v ----
#pragma unroll
    for (int i = 0; i < 4; ++i) {
#pragma unroll
        for (int j = 0; j < 4; ++j) {
            int col = bn + wn + j * 16 + (lane & 15);
            float bv = bias ? bias[col] : 0.f;
#pragma unroll
            for (int v = 0; v < 8; ++v) {
                int row = bm + wm + i * 16 + v + (lane >> 4) * 8;
                C[(size_t)row * N + col] = acc[i][j][v] + bv;
            }
        }
    }
}

// ---------------------------------------------------------------------------
// BatchNorm column statistics (biased var), 64 columns per block
// ---------------------------------------------------------------------------
__global__ __launch_bounds__(256)
void bn_stats_kernel(const float* __restrict__ Y, float* __restrict__ mean,
                     float* __restrict__ rstd, int rows, int D) {
    __shared__ float s1[256], s2[256];
    int col = blockIdx.x * 64 + (threadIdx.x & 63);
    int r0  = threadIdx.x >> 6;
    float s = 0.f, q = 0.f;
    for (int r = r0; r < rows; r += 4) {
        float x = Y[(size_t)r * D + col];
        s += x; q += x * x;
    }
    s1[threadIdx.x] = s; s2[threadIdx.x] = q;
    __syncthreads();
    if (threadIdx.x < 64) {
        float ts = s1[threadIdx.x] + s1[threadIdx.x + 64] + s1[threadIdx.x + 128] + s1[threadIdx.x + 192];
        float tq = s2[threadIdx.x] + s2[threadIdx.x + 64] + s2[threadIdx.x + 128] + s2[threadIdx.x + 192];
        float m = ts / rows;
        float v = tq / rows - m * m;
        mean[col] = m;
        rstd[col] = rsqrtf(v + BN_EPS);
    }
}

// ---------------------------------------------------------------------------
// Fused BN normalize -> ReLU -> bf16   (D = 2048, power of two)
// ---------------------------------------------------------------------------
__global__ __launch_bounds__(256)
void bn_relu_cvt_kernel(const float* __restrict__ Y,
                        const float* __restrict__ mean, const float* __restrict__ rstd,
                        const float* __restrict__ g, const float* __restrict__ be,
                        __bf16* __restrict__ out, int n) {
    int i = blockIdx.x * 256 + threadIdx.x;
    if (i >= n) return;
    int c = i & (D_DIM - 1);
    float x = (Y[i] - mean[c]) * rstd[c] * g[c] + be[c];
    out[i] = (__bf16)fmaxf(x, 0.f);
}

// ---------------------------------------------------------------------------
// Row L2 normalize -> bf16 (one block per row)
// ---------------------------------------------------------------------------
__global__ __launch_bounds__(256)
void l2norm_cvt_kernel(const float* __restrict__ F, __bf16* __restrict__ out, int D) {
    __shared__ float red[256];
    int row = blockIdx.x;
    const float* fr = F + (size_t)row * D;
    float s = 0.f;
    for (int j = threadIdx.x; j < D; j += 256) { float x = fr[j]; s += x * x; }
    red[threadIdx.x] = s;
    __syncthreads();
    for (int off = 128; off > 0; off >>= 1) {
        if (threadIdx.x < off) red[threadIdx.x] += red[threadIdx.x + off];
        __syncthreads();
    }
    float inv = 1.f / fmaxf(sqrtf(red[0]), 1e-12f);
    for (int j = threadIdx.x; j < D; j += 256)
        out[(size_t)row * D + j] = (__bf16)(fr[j] * inv);
}

// ---------------------------------------------------------------------------
// Row argmax of sim (diagonal masked), first-max tie break
// ---------------------------------------------------------------------------
__global__ __launch_bounds__(256)
void rowargmax_kernel(const float* __restrict__ S, int* __restrict__ nbr, int N) {
    __shared__ float bv[256];
    __shared__ int   bi[256];
    int row = blockIdx.x;
    const float* sr = S + (size_t)row * N;
    float best = -1e30f; int bidx = 0x7FFFFFFF;
    for (int j = threadIdx.x; j < N; j += 256) {
        float v = (j == row) ? -1e30f : sr[j];
        if (v > best || (v == best && j < bidx)) { best = v; bidx = j; }
    }
    bv[threadIdx.x] = best; bi[threadIdx.x] = bidx;
    __syncthreads();
    for (int off = 128; off > 0; off >>= 1) {
        if (threadIdx.x < off) {
            float ov = bv[threadIdx.x + off]; int oi = bi[threadIdx.x + off];
            if (ov > bv[threadIdx.x] || (ov == bv[threadIdx.x] && oi < bi[threadIdx.x])) {
                bv[threadIdx.x] = ov; bi[threadIdx.x] = oi;
            }
        }
        __syncthreads();
    }
    if (threadIdx.x == 0) nbr[row] = bi[0];
}

// ---------------------------------------------------------------------------
// Label propagation (connected components of top-1 neighbor graph)
// ---------------------------------------------------------------------------
__global__ void lab_init_kernel(int* lab, int n) {
    int i = blockIdx.x * 256 + threadIdx.x;
    if (i < n) lab[i] = i;
}
__global__ void lab_step1_kernel(const int* __restrict__ lab, const int* __restrict__ nbr,
                                 int* __restrict__ tmp, int n) {
    int i = blockIdx.x * 256 + threadIdx.x;
    if (i < n) tmp[i] = min(lab[i], lab[nbr[i]]);
}
__global__ void lab_step2_kernel(const int* __restrict__ lab, const int* __restrict__ nbr,
                                 int* __restrict__ tmp, int n) {
    int i = blockIdx.x * 256 + threadIdx.x;
    if (i < n) atomicMin(&tmp[nbr[i]], lab[i]);
}
__global__ void lab_step3_kernel(const int* __restrict__ tmp, int* __restrict__ lab, int n) {
    int i = blockIdx.x * 256 + threadIdx.x;
    if (i < n) { int t = tmp[i]; lab[i] = min(t, tmp[t]); }
}

// ---------------------------------------------------------------------------
// Masked contrastive loss: one block per row, atomicAdd scalar
// ---------------------------------------------------------------------------
__global__ void zero_out_kernel(float* out) { out[0] = 0.f; }

__global__ __launch_bounds__(256)
void loss_row_kernel(const float* __restrict__ S, const int* __restrict__ labels,
                     float* __restrict__ out, int N, float scale) {
    __shared__ float sE[256], sP[256], sC[256];
    int row = blockIdx.x;
    int myl = labels[row];
    const float* sr = S + (size_t)row * N;
    float se = 0.f, sp = 0.f, cnt = 0.f;
    for (int j = threadIdx.x; j < N; j += 256) {
        if (j == row) continue;
        float lg = sr[j] * INV_TEMP;
        se += expf(lg);
        if (labels[j] == myl) { sp += lg; cnt += 1.f; }
    }
    sE[threadIdx.x] = se; sP[threadIdx.x] = sp; sC[threadIdx.x] = cnt;
    __syncthreads();
    for (int off = 128; off > 0; off >>= 1) {
        if (threadIdx.x < off) {
            sE[threadIdx.x] += sE[threadIdx.x + off];
            sP[threadIdx.x] += sP[threadIdx.x + off];
            sC[threadIdx.x] += sC[threadIdx.x + off];
        }
        __syncthreads();
    }
    if (threadIdx.x == 0) {
        float logZ = logf(sE[0]);
        float mean_pos = (sP[0] - sC[0] * logZ) / sC[0];
        atomicAdd(out, -mean_pos * scale);
    }
}

// ---------------------------------------------------------------------------
// Host orchestration
// ---------------------------------------------------------------------------
extern "C" void kernel_launch(void* const* d_in, const int* in_sizes, int n_in,
                              void* d_out, int out_size, void* d_ws, size_t ws_size,
                              hipStream_t stream) {
    const int Bn = B_ROWS, D = D_DIM;
    const size_t BD = (size_t)Bn * D;
    const size_t DD = (size_t)D * D;
    const size_t BB = (size_t)Bn * Bn;

    // inputs: input_one, input_two, W1,b1,g1,be1, W2,b2,g2,be2, W3,b3
    const float* x1  = (const float*)d_in[0];
    const float* x2  = (const float*)d_in[1];
    const float* W1  = (const float*)d_in[2];
    const float* g1  = (const float*)d_in[4];
    const float* be1 = (const float*)d_in[5];
    const float* W2  = (const float*)d_in[6];
    const float* g2  = (const float*)d_in[8];
    const float* be2 = (const float*)d_in[9];
    const float* W3  = (const float*)d_in[10];
    const float* b3  = (const float*)d_in[11];
    // NOTE: b1/b2 are mathematically cancelled by the following BatchNorm.

    // workspace carve-out
    size_t off = 0;
    auto carve = [&](size_t bytes) -> void* {
        void* p = (char*)d_ws + off;
        off += (bytes + 255) & ~(size_t)255;
        return p;
    };
    __bf16* Wb1 = (__bf16*)carve(DD * 2);
    __bf16* Wb2 = (__bf16*)carve(DD * 2);
    __bf16* Wb3 = (__bf16*)carve(DD * 2);
    __bf16* Xb  = (__bf16*)carve(BD * 2);
    float*  Y   = (float*) carve(BD * 4);
    __bf16* F1b = (__bf16*)carve(BD * 2);
    __bf16* F2b = (__bf16*)carve(BD * 2);
    float*  S1  = (float*) carve(BB * 4);
    float*  S2  = (float*) carve(BB * 4);
    float*  mean = (float*)carve(D * 4);
    float*  rstd = (float*)carve(D * 4);
    int* nbr  = (int*)carve(Bn * 4);
    int* lab1 = (int*)carve(Bn * 4);
    int* lab2 = (int*)carve(Bn * 4);
    int* tmp  = (int*)carve(Bn * 4);
    (void)ws_size; (void)n_in; (void)in_sizes; (void)out_size;

    const int nBD = (int)BD, nDD = (int)DD;
    dim3 blk(256);
    dim3 gridBD((nBD + 255) / 256), gridDD((nDD + 255) / 256);
    dim3 gridLab((Bn + 255) / 256);

    // convert weights to bf16 (every call; no cached state allowed)
    cvt_bf16_kernel<<<gridDD, blk, 0, stream>>>(W1, Wb1, nDD);
    cvt_bf16_kernel<<<gridDD, blk, 0, stream>>>(W2, Wb2, nDD);
    cvt_bf16_kernel<<<gridDD, blk, 0, stream>>>(W3, Wb3, nDD);

    dim3 gemmGridProj(D / GBN, Bn / GBM);   // 8 x 32
    dim3 gemmGridSim(Bn / GBN, Bn / GBM);   // 16 x 32
    dim3 statsGrid(D / 64);

    auto project = [&](const float* x, __bf16* Fb) {
        cvt_bf16_kernel<<<gridBD, blk, 0, stream>>>(x, Xb, nBD);
        // layer 1
        gemm_nt_bf16_kernel<<<gemmGridProj, blk, 0, stream>>>(Xb, Wb1, Y, nullptr, Bn, D, D);
        bn_stats_kernel<<<statsGrid, blk, 0, stream>>>(Y, mean, rstd, Bn, D);
        bn_relu_cvt_kernel<<<gridBD, blk, 0, stream>>>(Y, mean, rstd, g1, be1, Xb, nBD);
        // layer 2
        gemm_nt_bf16_kernel<<<gemmGridProj, blk, 0, stream>>>(Xb, Wb2, Y, nullptr, Bn, D, D);
        bn_stats_kernel<<<statsGrid, blk, 0, stream>>>(Y, mean, rstd, Bn, D);
        bn_relu_cvt_kernel<<<gridBD, blk, 0, stream>>>(Y, mean, rstd, g2, be2, Xb, nBD);
        // layer 3 (bias matters here)
        gemm_nt_bf16_kernel<<<gemmGridProj, blk, 0, stream>>>(Xb, Wb3, Y, b3, Bn, D, D);
        // l2 normalize -> bf16 features
        l2norm_cvt_kernel<<<dim3(Bn), blk, 0, stream>>>(Y, Fb, D);
    };

    project(x1, F1b);
    project(x2, F2b);

    // similarity matrices
    gemm_nt_bf16_kernel<<<gemmGridSim, blk, 0, stream>>>(F1b, F1b, S1, nullptr, Bn, Bn, D);
    gemm_nt_bf16_kernel<<<gemmGridSim, blk, 0, stream>>>(F2b, F2b, S2, nullptr, Bn, Bn, D);

    // cluster labels via top-1 neighbor + min label propagation
    auto components = [&](const float* S, int* lab) {
        rowargmax_kernel<<<dim3(Bn), blk, 0, stream>>>(S, nbr, Bn);
        lab_init_kernel<<<gridLab, blk, 0, stream>>>(lab, Bn);
        for (int it = 0; it < 24; ++it) {
            lab_step1_kernel<<<gridLab, blk, 0, stream>>>(lab, nbr, tmp, Bn);
            lab_step2_kernel<<<gridLab, blk, 0, stream>>>(lab, nbr, tmp, Bn);
            lab_step3_kernel<<<gridLab, blk, 0, stream>>>(tmp, lab, Bn);
        }
    };
    components(S1, lab1);
    components(S2, lab2);

    // loss = 0.5 * (L(s1/T, m2) + L(s2/T, m1));  per-row scale = 1/(2B)
    float scale = 1.0f / (2.0f * (float)Bn);
    zero_out_kernel<<<1, 1, 0, stream>>>((float*)d_out);
    loss_row_kernel<<<dim3(Bn), blk, 0, stream>>>(S1, lab2, (float*)d_out, Bn, scale);
    loss_row_kernel<<<dim3(Bn), blk, 0, stream>>>(S2, lab1, (float*)d_out, Bn, scale);
}
